// MHSA_51651276702217
// MI455X (gfx1250) — matrix operand order repaired and checked by reading, not compiled, not run on hardware
//
#include <hip/hip_runtime.h>

// MI455X / gfx1250, wave32. All matmuls via v_wmma_f32_16x16x32_bf16.
using bf16 = __bf16;
typedef __attribute__((ext_vector_type(16))) __bf16 v16bf;
typedef __attribute__((ext_vector_type(8)))  float  v8f;

struct alignas(16) U4 { unsigned int x, y, z, w; };
union V16 { v16bf v; U4 u[2]; __bf16 e[16]; };

__device__ inline v16bf load_frag16(const bf16* p0, const bf16* p1) {
  V16 r;
  r.u[0] = *(const U4*)p0;
  r.u[1] = *(const U4*)p1;
  return r.v;
}

__device__ inline v8f wmma_bf16(v16bf a, v16bf b, v8f c) {
  return __builtin_amdgcn_wmma_f32_16x16x32_bf16(false, a, false, b, (short)0, c,
                                                 false, false);
}

__device__ inline v8f zero8() { v8f z = {0.f,0.f,0.f,0.f,0.f,0.f,0.f,0.f}; return z; }

// ---- CDNA5 async global->LDS copy (ASYNCcnt path), with sync fallback -----
#if defined(__has_builtin)
#if __has_builtin(__builtin_amdgcn_global_load_async_to_lds_b128)
#define HAVE_ASYNC_LDS 1
#endif
#endif

typedef int vsi4 __attribute__((vector_size(16)));
typedef __attribute__((address_space(1))) vsi4* as1_v4i;
typedef __attribute__((address_space(3))) vsi4* as3_v4i;

__device__ inline void copy16_g2l(const bf16* g, bf16* l) {
#ifdef HAVE_ASYNC_LDS
  __builtin_amdgcn_global_load_async_to_lds_b128(
      (as1_v4i)(unsigned long long)g,
      (as3_v4i)(unsigned)(unsigned long long)l,
      0, 0);
#else
  *(U4*)l = *(const U4*)g;
#endif
}
__device__ inline void async_commit_wait() {
#ifdef HAVE_ASYNC_LDS
  asm volatile("s_wait_asynccnt 0x0" ::: "memory");
#endif
}

// ---------------------------------------------------------------- converts
__global__ void cvt_f32_bf16(const float* __restrict__ src, bf16* __restrict__ dst, int n) {
  int i = blockIdx.x * blockDim.x + threadIdx.x;
  if (i < n) dst[i] = (bf16)src[i];
}

// Pack wq|wk|wv transposed: wt[n][k] = w(k,n), n in [0,2304), k in [0,768).
__global__ void pack_wqkvT_kernel(const float* __restrict__ wq,
                                  const float* __restrict__ wk,
                                  const float* __restrict__ wv,
                                  bf16* __restrict__ wt) {
  int i = blockIdx.x * blockDim.x + threadIdx.x;
  if (i >= 2304 * 768) return;
  int n = i / 768, k = i % 768;
  const float* src = (n < 768) ? wq : (n < 1536 ? wk : wv);
  wt[i] = (bf16)src[k * 768 + (n % 768)];
}

// wo transposed: wt[n][k] = wo[k][n], both 768.
__global__ void pack_woT_kernel(const float* __restrict__ wo, bf16* __restrict__ wt) {
  int i = blockIdx.x * blockDim.x + threadIdx.x;
  if (i >= 768 * 768) return;
  int n = i / 768, k = i % 768;
  wt[i] = (bf16)wo[k * 768 + n];
}

// ---------------------------------------------------------------- GEMM core
// 256 threads = 8 waves as 4(M)x2(N); WG tile 128x128; wave tile 32x64;
// K-step 32, K=768 (24 steps), double-buffered LDS, async prefetch.
// X: [M,768] row-major. WT: [N,768] row-major (pre-transposed weights) so
// both LDS tiles have identical shape and b128-only staging.
__device__ inline void gemm_pipe(const bf16* __restrict__ Xt,   // + m0*768
                                 const bf16* __restrict__ WTt,  // + n0*768
                                 bf16 As[2][128][40], bf16 Bs[2][128][40],
                                 v8f acc[2][4]) {
  int tid = threadIdx.x, lane = tid & 31, wave = tid >> 5;
  int wm = wave >> 1, wn = wave & 1, half = lane >> 4, l16 = lane & 15;
  int srow = tid >> 1, scol = (tid & 1) * 16;  // 128x32 tile: 16 elems/thread
#pragma unroll
  for (int ri = 0; ri < 2; ++ri)
#pragma unroll
    for (int t = 0; t < 4; ++t) acc[ri][t] = zero8();

  const int KSTEPS = 24;
  // prologue: stage tile 0 into buffer 0
  copy16_g2l(&Xt[srow * 768 + scol],      &As[0][srow][scol]);
  copy16_g2l(&Xt[srow * 768 + scol + 8],  &As[0][srow][scol + 8]);
  copy16_g2l(&WTt[srow * 768 + scol],     &Bs[0][srow][scol]);
  copy16_g2l(&WTt[srow * 768 + scol + 8], &Bs[0][srow][scol + 8]);
  async_commit_wait();
  __syncthreads();

  for (int kk = 0; kk < KSTEPS; ++kk) {
    int p = kk & 1;
    if (kk + 1 < KSTEPS) {
      int k0 = (kk + 1) * 32;
      copy16_g2l(&Xt[srow * 768 + k0 + scol],      &As[1 - p][srow][scol]);
      copy16_g2l(&Xt[srow * 768 + k0 + scol + 8],  &As[1 - p][srow][scol + 8]);
      copy16_g2l(&WTt[srow * 768 + k0 + scol],     &Bs[1 - p][srow][scol]);
      copy16_g2l(&WTt[srow * 768 + k0 + scol + 8], &Bs[1 - p][srow][scol + 8]);
    }
    int am0 = wm * 32 + l16;
    v16bf af0 = load_frag16(&As[p][am0][half * 8],      &As[p][am0][16 + half * 8]);
    v16bf af1 = load_frag16(&As[p][am0 + 16][half * 8], &As[p][am0 + 16][16 + half * 8]);
#pragma unroll
    for (int t = 0; t < 4; ++t) {
      int bn = wn * 64 + t * 16 + l16;
      v16bf bw = load_frag16(&Bs[p][bn][half * 8], &Bs[p][bn][16 + half * 8]);
      acc[0][t] = wmma_bf16(af0, bw, acc[0][t]);
      acc[1][t] = wmma_bf16(af1, bw, acc[1][t]);
    }
    async_commit_wait();
    __syncthreads();
  }
}

// ------------------------------------------------- QKV projection (fused)
// C[4096,2304] = X@Wqkv + bias. Q,K -> [B,H,N,64] (Q pre-scaled by 1/8);
// V -> [B,H,64,N] (transposed) so attention B-fragments are contiguous.
__global__ __launch_bounds__(256) void gemm_qkv_kernel(
    const bf16* __restrict__ X, const bf16* __restrict__ WT,
    const float* __restrict__ bq, const float* __restrict__ bk,
    const float* __restrict__ bvv,
    bf16* __restrict__ Qo, bf16* __restrict__ Ko, bf16* __restrict__ Vo) {
  __shared__ bf16 As[2][128][40];
  __shared__ bf16 Bs[2][128][40];
  int mt = blockIdx.x % 32, nt = blockIdx.x / 32;
  int m0 = mt * 128, n0 = nt * 128;
  v8f acc[2][4];
  gemm_pipe(X + m0 * 768, WT + n0 * 768, As, Bs, acc);

  int lane = threadIdx.x & 31, wave = threadIdx.x >> 5;
  int wm = wave >> 1, wn = wave & 1, half = lane >> 4, l16 = lane & 15;
#pragma unroll
  for (int ri = 0; ri < 2; ++ri)
#pragma unroll
    for (int t = 0; t < 4; ++t)
#pragma unroll
      for (int r = 0; r < 8; ++r) {
        int m = m0 + wm * 32 + ri * 16 + half * 8 + r;
        int n = n0 + wn * 64 + t * 16 + l16;
        int bb = m >> 11, nq = m & 2047;
        int region = n / 768, nn = n % 768;
        int h = nn >> 6, d = nn & 63;
        float a = acc[ri][t][r];
        if (region == 0) {
          Qo[((bb * 12 + h) * 2048 + nq) * 64 + d] = (bf16)((a + bq[nn]) * 0.125f);
        } else if (region == 1) {
          Ko[((bb * 12 + h) * 2048 + nq) * 64 + d] = (bf16)(a + bk[nn]);
        } else {
          Vo[((bb * 12 + h) * 64 + d) * 2048 + nq] = (bf16)(a + bvv[nn]);
        }
      }
}

// ---------------------------------------------------- flash attention core
// Grid: (b,h,qblock) = 24*32 = 768 blocks, 128 threads = 4 waves x 16 rows.
// Barrier-free: Q/K/V fragments straight from HBM (layouts arranged for it);
// only LDS use is the per-wave P transpose (same-wave, in-order DS).
__global__ __launch_bounds__(128) void attn_kernel(
    const bf16* __restrict__ Q, const bf16* __restrict__ K,
    const bf16* __restrict__ VT, bf16* __restrict__ Z) {
  __shared__ bf16 Pb[4][16][72];  // per-wave P transpose staging

  const int N = 2048, D = 64;
  int qb = blockIdx.x & 31;
  int bh = blockIdx.x >> 5;  // 0..23
  int b = bh / 12, h = bh % 12;

  int tid = threadIdx.x, lane = tid & 31, wave = tid >> 5;
  int half = lane >> 4, l16 = lane & 15;

  const bf16* Qb_ = Q + (long)bh * N * D;
  const bf16* Kb_ = K + (long)bh * N * D;
  const bf16* Vt_ = VT + (long)bh * D * N;  // [64 dims][2048 keys]

  int qrow = qb * 64 + wave * 16 + l16;
  v16bf qf[2];
#pragma unroll
  for (int s = 0; s < 2; ++s)
    qf[s] = load_frag16(&Qb_[qrow * D + s * 32 + half * 8],
                        &Qb_[qrow * D + s * 32 + 16 + half * 8]);

  float mrow[8], lrow[8];
  v8f o[4];
#pragma unroll
  for (int r = 0; r < 8; ++r) { mrow[r] = -3.0e38f; lrow[r] = 0.f; }
#pragma unroll
  for (int t = 0; t < 4; ++t) o[t] = zero8();

  for (int kb = 0; kb < 32; ++kb) {
    int kbase = kb * 64;

    // S = Q*K^T (16x64), K fragments contiguous from HBM.
    v8f s4[4];
#pragma unroll
    for (int t = 0; t < 4; ++t) s4[t] = zero8();
#pragma unroll
    for (int t = 0; t < 4; ++t) {
      int key = kbase + t * 16 + l16;
#pragma unroll
      for (int ks = 0; ks < 2; ++ks) {
        v16bf kf = load_frag16(&Kb_[key * D + ks * 32 + half * 8],
                               &Kb_[key * D + ks * 32 + 16 + half * 8]);
        s4[t] = wmma_bf16(qf[ks], kf, s4[t]);
      }
    }

    // Online softmax; row = (r + 8*half), cols live in 16 lanes of a half.
    float corr[8];
#pragma unroll
    for (int r = 0; r < 8; ++r) {
      float v = fmaxf(fmaxf(s4[0][r], s4[1][r]), fmaxf(s4[2][r], s4[3][r]));
      v = fmaxf(v, __shfl_xor(v, 1, 32));
      v = fmaxf(v, __shfl_xor(v, 2, 32));
      v = fmaxf(v, __shfl_xor(v, 4, 32));
      v = fmaxf(v, __shfl_xor(v, 8, 32));
      float nm = fmaxf(mrow[r], v);
      corr[r] = __expf(mrow[r] - nm);
      mrow[r] = nm;
    }
    float pf[4][8];
#pragma unroll
    for (int t = 0; t < 4; ++t)
#pragma unroll
      for (int r = 0; r < 8; ++r) {
        float p = __expf(s4[t][r] - mrow[r]);
        pf[t][r] = p;
        Pb[wave][half * 8 + r][t * 16 + l16] = (bf16)p;  // transpose via LDS
      }
#pragma unroll
    for (int r = 0; r < 8; ++r) {
      float rs = pf[0][r] + pf[1][r] + pf[2][r] + pf[3][r];
      rs += __shfl_xor(rs, 1, 32);
      rs += __shfl_xor(rs, 2, 32);
      rs += __shfl_xor(rs, 4, 32);
      rs += __shfl_xor(rs, 8, 32);
      lrow[r] = lrow[r] * corr[r] + rs;
#pragma unroll
      for (int t = 0; t < 4; ++t) o[t][r] *= corr[r];
    }
    asm volatile("" ::: "memory");  // keep P stores before P loads (same wave)

    // O += P*V; P as A from LDS; V^T as B straight from HBM ([H,D,N] layout).
#pragma unroll
    for (int ks = 0; ks < 2; ++ks) {
      v16bf pa = load_frag16(&Pb[wave][l16][ks * 32 + half * 8],
                             &Pb[wave][l16][ks * 32 + 16 + half * 8]);
#pragma unroll
      for (int t = 0; t < 4; ++t) {
        const bf16* vp = &Vt_[(t * 16 + l16) * N + kbase + ks * 32 + half * 8];
        v16bf vf = load_frag16(vp, vp + 16);
        o[t] = wmma_bf16(pa, vf, o[t]);
      }
    }
    asm volatile("" ::: "memory");  // P loads done before next-iter stores
  }

  // Normalize, merge heads: Z[b*N + row][h*64 + dim] (bf16, feeds out-proj).
  long zrow = (long)b * N + qb * 64 + wave * 16;
#pragma unroll
  for (int t = 0; t < 4; ++t)
#pragma unroll
    for (int r = 0; r < 8; ++r) {
      int m = half * 8 + r;
      float val = o[t][r] / lrow[r];
      Z[(zrow + m) * 768 + h * 64 + t * 16 + l16] = (bf16)val;
    }
}

// ------------------------------------------------------- output projection
__global__ __launch_bounds__(256) void gemm_out_kernel(
    const bf16* __restrict__ Zin, const bf16* __restrict__ WT,
    const float* __restrict__ bo, float* __restrict__ out) {
  __shared__ bf16 As[2][128][40];
  __shared__ bf16 Bs[2][128][40];
  int mt = blockIdx.x % 32, nt = blockIdx.x / 32;
  int m0 = mt * 128, n0 = nt * 128;
  v8f acc[2][4];
  gemm_pipe(Zin + m0 * 768, WT + n0 * 768, As, Bs, acc);

  int lane = threadIdx.x & 31, wave = threadIdx.x >> 5;
  int wm = wave >> 1, wn = wave & 1, half = lane >> 4, l16 = lane & 15;
#pragma unroll
  for (int ri = 0; ri < 2; ++ri)
#pragma unroll
    for (int t = 0; t < 4; ++t)
#pragma unroll
      for (int r = 0; r < 8; ++r) {
        int m = m0 + wm * 32 + ri * 16 + half * 8 + r;
        int n = n0 + wn * 64 + t * 16 + l16;
        out[m * 768 + n] = acc[ri][t][r] + bo[n];
      }
}

// ------------------------------------------------------------------ launch
extern "C" void kernel_launch(void* const* d_in, const int* in_sizes, int n_in,
                              void* d_out, int out_size, void* d_ws, size_t ws_size,
                              hipStream_t stream) {
  const float* x  = (const float*)d_in[0];
  const float* wq = (const float*)d_in[1];
  const float* bq = (const float*)d_in[2];
  const float* wk = (const float*)d_in[3];
  const float* bk = (const float*)d_in[4];
  const float* wv = (const float*)d_in[5];
  const float* bv = (const float*)d_in[6];
  const float* wo = (const float*)d_in[7];
  const float* bo = (const float*)d_in[8];
  float* out = (float*)d_out;

  char* ws = (char*)d_ws;
  bf16* xb    = (bf16*)(ws);              // x bf16            [4096,768]
  bf16* wqkvT = (bf16*)(ws +  6291456);   // (wq|wk|wv)^T bf16 [2304,768]
  bf16* woT   = (bf16*)(ws +  9830400);   // wo^T bf16         [768,768]
  bf16* Qw    = (bf16*)(ws + 11010048);   // Q [B,H,N,64] bf16 (pre-scaled)
  bf16* Kw    = (bf16*)(ws + 17301504);   // K [B,H,N,64] bf16
  bf16* VwT   = (bf16*)(ws + 23592960);   // V [B,H,64,N] bf16 (transposed)
  bf16* Zw    = (bf16*)(ws + 29884416);   // attn out [4096,768] (ends 36175872)

  cvt_f32_bf16<<<(3145728 + 255) / 256, 256, 0, stream>>>(x, xb, 3145728);
  pack_wqkvT_kernel<<<(1769472 + 255) / 256, 256, 0, stream>>>(wq, wk, wv, wqkvT);
  pack_woT_kernel<<<(589824 + 255) / 256, 256, 0, stream>>>(wo, woT);

  gemm_qkv_kernel<<<32 * 18, 256, 0, stream>>>(xb, wqkvT, bq, bk, bv, Qw, Kw, VwT);
  attn_kernel<<<768, 128, 0, stream>>>(Qw, Kw, VwT, Zw);
  gemm_out_kernel<<<32 * 6, 256, 0, stream>>>(Zw, woT, bo, out);
}
